// PointFeatureAttention_52364241273451
// MI455X (gfx1250) — compile-verified, matrix operand order checked by reading, not run on hardware
//
#include <hip/hip_runtime.h>
#include <hip/hip_fp16.h>

// ---------------------------------------------------------------------------
// PointFeatureAttention fused forward for MI455X (gfx1250, wave32, WMMA).
// B=2, R=4096, K=8, M=4, F=128 -> BR=8192 rays, L=32 tokens/ray.
// Compute-bound (~150 GFLOP GEMM vs ~140MB traffic) => fp16 WMMA w/ f32 accum.
// keys@Wka and values@Wva are folded into Wkey/Wval (no ReLU between),
// out_proj@Wout folded likewise.
// ---------------------------------------------------------------------------

typedef _Float16 v16h __attribute__((ext_vector_type(16)));
typedef float    v8f  __attribute__((ext_vector_type(8)));
typedef unsigned u32x4 __attribute__((ext_vector_type(4)));

#define N_RAYS      8192
#define OUT_ATTN_OFF 1048576u   // B*R*128 floats, attn weights follow

// ---- workspace layout (bytes) ----
#define OFF_QS     0u                        // qh per ray, f32 [8192*128]
#define OFF_WK0F   4194304u                  // layer1 key frags  (48 KB)
#define OFF_WV0F   (OFF_WK0F + 49152u)
#define OFF_WK1F   (OFF_WV0F + 49152u)       // layer2 frags (128 KB each)
#define OFF_WV1F   (OFF_WK1F + 131072u)
#define OFF_WCKF   (OFF_WV1F + 131072u)      // folded layer3 frags (64 KB each)
#define OFF_WCVF   (OFF_WCKF + 65536u)
#define OFF_WCK32  (OFF_WCVF + 65536u)       // folded f32 mats
#define OFF_WCV32  (OFF_WCK32 + 131072u)
#define OFF_WOO    (OFF_WCV32 + 131072u)
#define OFF_BCK    (OFF_WOO + 65536u)
#define OFF_BCV    (OFF_BCK + 512u)
#define OFF_BOC    (OFF_BCV + 512u)

// ===========================================================================
// Fold Wck = Wkey@Wka (+bias), Wcv = Wval@Wva, Woo = Wo@Wout
// ===========================================================================
__global__ void k_combine(const float* __restrict__ Wkey, const float* __restrict__ bkey,
                          const float* __restrict__ Wka,  const float* __restrict__ bka,
                          const float* __restrict__ Wval, const float* __restrict__ bval,
                          const float* __restrict__ Wva,  const float* __restrict__ bva,
                          const float* __restrict__ Wo,   const float* __restrict__ bo,
                          const float* __restrict__ Wout, const float* __restrict__ bout,
                          char* __restrict__ ws) {
  int idx = blockIdx.x * 256 + threadIdx.x;
  float* wck = (float*)(ws + OFF_WCK32);
  float* wcv = (float*)(ws + OFF_WCV32);
  float* woo = (float*)(ws + OFF_WOO);
  float* bck = (float*)(ws + OFF_BCK);
  float* bcv = (float*)(ws + OFF_BCV);
  float* boc = (float*)(ws + OFF_BOC);
  if (idx < 32768) {                       // Wck[256][128]
    int k = idx >> 7, n = idx & 127; float a = 0.f;
    for (int j = 0; j < 128; ++j) a += Wkey[k*128 + j] * Wka[j*128 + n];
    wck[idx] = a;
  } else if (idx < 65536) {                // Wcv[256][128]
    int i = idx - 32768; int k = i >> 7, n = i & 127; float a = 0.f;
    for (int j = 0; j < 128; ++j) a += Wval[k*128 + j] * Wva[j*128 + n];
    wcv[i] = a;
  } else if (idx < 81920) {                // Woo[128][128]
    int i = idx - 65536; int e = i >> 7, n = i & 127; float a = 0.f;
    for (int j = 0; j < 128; ++j) a += Wo[e*128 + j] * Wout[j*128 + n];
    woo[i] = a;
  } else if (idx < 82048) {                // bck
    int n = idx - 81920; float a = bka[n];
    for (int j = 0; j < 128; ++j) a += bkey[j] * Wka[j*128 + n];
    bck[n] = a;
  } else if (idx < 82176) {                // bcv
    int n = idx - 82048; float a = bva[n];
    for (int j = 0; j < 128; ++j) a += bval[j] * Wva[j*128 + n];
    bcv[n] = a;
  } else if (idx < 82304) {                // boc
    int n = idx - 82176; float a = bout[n];
    for (int j = 0; j < 128; ++j) a += bo[j] * Wout[j*128 + n];
    boc[n] = a;
  }
}

// ===========================================================================
// Pack weights into f16 WMMA-B fragment-major layout.
// Fragment (kt,nt) = B[kt*32..+32)[nt*16..+16); per-lane 16 halves stored
// contiguously (32 B) so the consumer issues ds_load_b128 pairs.
// Assumed B layout mirrors the documented A layout: lane L holds column
// n = L%16; K-pairs K0(v, L/16) with K0 = (v<4) ? 8*half+2v : 16+8*half+2(v-4).
// ===========================================================================
__global__ void k_pack(const float* __restrict__ Wk0, const float* __restrict__ Wv0,
                       const float* __restrict__ Wk1, const float* __restrict__ Wv1,
                       char* __restrict__ ws) {
  const float* src; _Float16* dst; int Kv, N, NT, fl;
  int f = blockIdx.x;
  const float* wck32 = (const float*)(ws + OFF_WCK32);
  const float* wcv32 = (const float*)(ws + OFF_WCV32);
  if (f < 48)       { src = Wk0;   Kv = 89;  N = 256; NT = 16; fl = f;       dst = (_Float16*)(ws + OFF_WK0F); }
  else if (f < 96)  { src = Wv0;   Kv = 89;  N = 256; NT = 16; fl = f - 48;  dst = (_Float16*)(ws + OFF_WV0F); }
  else if (f < 224) { src = Wk1;   Kv = 256; N = 256; NT = 16; fl = f - 96;  dst = (_Float16*)(ws + OFF_WK1F); }
  else if (f < 352) { src = Wv1;   Kv = 256; N = 256; NT = 16; fl = f - 224; dst = (_Float16*)(ws + OFF_WV1F); }
  else if (f < 416) { src = wck32; Kv = 256; N = 128; NT = 8;  fl = f - 352; dst = (_Float16*)(ws + OFF_WCKF); }
  else              { src = wcv32; Kv = 256; N = 128; NT = 8;  fl = f - 416; dst = (_Float16*)(ws + OFF_WCVF); }
  int p = threadIdx.x;                 // one K-pair per thread (256 = 32 lanes * 8 pairs)
  int L = p >> 3, v = p & 7;
  int kt = fl / NT, nt = fl - kt * NT;
  int half = L >> 4, n = nt * 16 + (L & 15);
  int K0 = (v < 4) ? (half * 8 + 2 * v) : (16 + half * 8 + 2 * (v - 4));
  int k0 = kt * 32 + K0;
  float f0 = (k0     < Kv) ? src[(size_t)k0 * N + n]       : 0.f;
  float f1 = (k0 + 1 < Kv) ? src[(size_t)(k0 + 1) * N + n] : 0.f;
  _Float16* d = dst + (size_t)fl * 512 + L * 16 + v * 2;
  d[0] = (_Float16)f0; d[1] = (_Float16)f1;
}

// ===========================================================================
// Per-ray query: qh = ((pos_enc(dir)@Wr0+br0)@Wqr+bqr)@Wq+bq, scaled 1/sqrt(16)
// ===========================================================================
__global__ void k_query(const float* __restrict__ dirs,
                        const float* __restrict__ Wr0, const float* __restrict__ br0,
                        const float* __restrict__ Wqr, const float* __restrict__ bqr,
                        const float* __restrict__ Wq,  const float* __restrict__ bq,
                        char* __restrict__ ws) {
  __shared__ float enc[27];
  __shared__ float t64[64];
  __shared__ float t128[128];
  int r = blockIdx.x, tid = threadIdx.x;
  if (tid < 27) {
    if (tid < 3) enc[tid] = dirs[r * 3 + tid];
    else {
      int j = tid - 3; int d = j >> 3, rem = j & 7;
      float x = dirs[r * 3 + d] * (float)(1 << (rem & 3));
      enc[tid] = (rem < 4) ? __sinf(x) : __cosf(x);
    }
  }
  __syncthreads();
  if (tid < 64) {
    float a = br0[tid];
    for (int j = 0; j < 27; ++j) a += enc[j] * Wr0[j * 64 + tid];
    t64[tid] = a;
  }
  __syncthreads();
  { float a = bqr[tid];
    for (int j = 0; j < 64; ++j) a += t64[j] * Wqr[j * 128 + tid];
    t128[tid] = a; }
  __syncthreads();
  { float a = bq[tid];
    for (int j = 0; j < 128; ++j) a += t128[j] * Wq[j * 128 + tid];
    ((float*)(ws + OFF_QS))[(size_t)r * 128 + tid] = a * 0.25f; }
}

// ===========================================================================
// WMMA fragment loaders (documented CDNA5 16-bit layouts)
// ===========================================================================
__device__ __forceinline__ v16h load_afrag(const _Float16* act, int S, int mt, int kbase, int lane) {
  int row  = mt * 16 + (lane & 15);
  int half = lane >> 4;
  union { v16h v; unsigned u[8]; } a;
#pragma unroll
  for (int t = 0; t < 8; ++t) {
    int K0 = (t < 4) ? (half * 8 + 2 * t) : (16 + half * 8 + 2 * (t - 4));
    a.u[t] = *reinterpret_cast<const unsigned*>(act + (size_t)row * S + kbase + K0);
  }
  return a.v;
}

__device__ __forceinline__ v16h load_bfrag(const _Float16* wbuf, int fid, int lane) {
  union { v16h v; u32x4 q[2]; } b;
  const u32x4* p = reinterpret_cast<const u32x4*>(wbuf + (size_t)fid * 512 + lane * 16);
  b.q[0] = p[0]; b.q[1] = p[1];
  return b.v;
}

// ===========================================================================
// One MLP layer: [128 x KT*32] @ [KT*32 x NT*16] + bias (+ReLU) -> f16 LDS.
// Weights staged from fragment-packed global into 64 KB LDS in K-chunks;
// 8 waves each own one 16-row stripe, accumulators persist across chunks.
// ===========================================================================
template<int KT, int NT, bool RELU>
__device__ __forceinline__ void gemm_layer(const _Float16* __restrict__ actIn, int Sin,
                                           const _Float16* __restrict__ wfragG,
                                           _Float16* __restrict__ wbuf,
                                           const float* __restrict__ biasS,
                                           _Float16* __restrict__ actOut, int Sout, int tid) {
  const int wid = tid >> 5, lane = tid & 31;
  constexpr int CH_KT = ((64 / NT) < KT) ? (64 / NT) : KT;   // <=64 frags staged at once
  v8f acc[NT];
#pragma unroll
  for (int nt = 0; nt < NT; ++nt)
#pragma unroll
    for (int q = 0; q < 8; ++q) acc[nt][q] = 0.f;

  for (int kb = 0; kb < KT; kb += CH_KT) {
    int nk = ((KT - kb) < CH_KT) ? (KT - kb) : CH_KT;
    __syncthreads();
    {   // stage nk*NT fragments (contiguous) into LDS
      const u32x4* s = (const u32x4*)(wfragG + (size_t)kb * NT * 512);
      u32x4* d = (u32x4*)wbuf;
      int tot = nk * NT * 64;                 // 64 x u32x4 per fragment
      for (int i = tid; i < tot; i += 256) d[i] = s[i];
    }
    if (kb + CH_KT < KT)                       // gfx1250 global_prefetch of next chunk
      __builtin_prefetch(wfragG + (size_t)(kb + CH_KT) * NT * 512, 0, 1);
    __syncthreads();
#pragma unroll
    for (int k2 = 0; k2 < CH_KT; ++k2) {
      if (kb + k2 >= KT) break;
      v16h a = load_afrag(actIn, Sin, wid, (kb + k2) * 32, lane);
#pragma unroll
      for (int nt = 0; nt < NT; ++nt) {
        v16h b = load_bfrag(wbuf, k2 * NT + nt, lane);
        acc[nt] = __builtin_amdgcn_wmma_f32_16x16x32_f16(false, a, false, b,
                                                         (short)0, acc[nt], false, false);
      }
    }
  }
  const int nloc = lane & 15, hg = lane >> 4;  // D layout: m = r + 8*(lane/16)
#pragma unroll
  for (int nt = 0; nt < NT; ++nt) {
    float bb = biasS[nt * 16 + nloc];
#pragma unroll
    for (int r = 0; r < 8; ++r) {
      float v = acc[nt][r] + bb;
      if (RELU) v = fmaxf(v, 0.f);
      actOut[(size_t)(wid * 16 + hg * 8 + r) * Sout + nt * 16 + nloc] = (_Float16)v;
    }
  }
}

// build 128x96 f16 input tile: [even|odd] feature split + 25 pos-enc + zero pad
__device__ __forceinline__ void buildIN(_Float16* __restrict__ IN,
                                        const float* __restrict__ feats,
                                        const float* __restrict__ PE,
                                        int gr0, int parity, int tid) {
  for (int idx = tid; idx < 128 * 96; idx += 256) {
    int t = idx / 96, c = idx - t * 96;
    int rr = t >> 5, tok = t & 31;
    float v;
    if (c < 64) {
      int kk = tok >> 2, mm = tok & 3;
      v = feats[((size_t)((gr0 + rr) * 8 + kk) * 4 + mm) * 128 + 2 * c + parity];
    } else if (c < 89) {
      v = PE[(rr * 8 + (tok >> 2)) * 25 + (c - 64)];
    } else v = 0.f;
    IN[idx] = (_Float16)v;
  }
}

// ===========================================================================
// Main fused kernel: 1 block = 4 rays = 128 tokens, 256 threads = 8 waves.
// ===========================================================================
#define SMEM_MAIN 307456

__global__ __launch_bounds__(256) void k_main(
    const float* __restrict__ features,
    const float* __restrict__ distance, const float* __restrict__ pitch,
    const float* __restrict__ azimuth,  const float* __restrict__ tx_distance,
    const float* __restrict__ tx_azimuth, const float* __restrict__ tx_elevation,
    const float* __restrict__ bk0, const float* __restrict__ bk1,
    const float* __restrict__ bv0, const float* __restrict__ bv1,
    const char* __restrict__ ws, float* __restrict__ out) {
  extern __shared__ char smem[];
  _Float16* IN   = (_Float16*)smem;          // 128x96
  _Float16* H1   = IN + 128 * 96;            // 128x256
  _Float16* H2   = H1 + 128 * 256;           // 128x256
  _Float16* KH   = H2 + 128 * 256;           // 128x128
  _Float16* VH   = KH + 128 * 128;           // 128x128
  _Float16* WBUF = VH + 128 * 128;           // 32768 halves = 64 KB stage
  float* PK    = (float*)(WBUF + 32768);     // 32x25
  float* PV    = PK + 800;                   // 32x25
  float* BIASS = PV + 800;                   // 1536
  float* QSH   = BIASS + 1536;               // 4x128
  float* ATTN  = QSH + 512;                  // 4x8x32
  float* OUTH  = ATTN + 1024;                // 4x128

  const int tid = threadIdx.x;
  const int gr0 = blockIdx.x * 4;

  const float* qs  = (const float*)(ws + OFF_QS);
  const float* bck = (const float*)(ws + OFF_BCK);
  const float* bcv = (const float*)(ws + OFF_BCV);
  const float* boc = (const float*)(ws + OFF_BOC);
  const float* woo = (const float*)(ws + OFF_WOO);

  // ---- stage biases / queries ----
  BIASS[tid]        = bk0[tid];
  BIASS[256 + tid]  = bk1[tid];
  BIASS[640 + tid]  = bv0[tid];
  BIASS[896 + tid]  = bv1[tid];
  if (tid < 128) {
    BIASS[512 + tid]  = bck[tid];
    BIASS[1152 + tid] = bcv[tid];
    BIASS[1280 + tid] = boc[tid];
  }
  QSH[tid]       = qs[(size_t)gr0 * 128 + tid];
  QSH[256 + tid] = qs[(size_t)gr0 * 128 + 256 + tid];

  // ---- geometric encodings, one (ray,k) per lane ----
  if (tid < 32) {
    int rr = tid >> 3, kk = tid & 7;
    int g = (gr0 + rr) * 8 + kk;
    float d  = distance[g], p = pitch[g], az = azimuth[g];
    float td = tx_distance[g];
    td = (td < 0.01f) ? 100.f : (1.f / td);
    float ta = tx_azimuth[g], te = tx_elevation[g];
    float* ek = PK + tid * 25;
    float* ev = PV + tid * 25;
    ek[0] = d; ev[0] = td;
#pragma unroll
    for (int i = 0; i < 4; ++i) {
      float fs = (float)(1 << i);
      ek[1 + i]  = __sinf(d * fs);  ek[5 + i]  = __cosf(d * fs);
      ek[9 + i]  = __sinf(p * fs);  ek[13 + i] = __cosf(p * fs);
      ek[17 + i] = __sinf(az * fs); ek[21 + i] = __cosf(az * fs);
      ev[1 + i]  = __sinf(td * fs); ev[5 + i]  = __cosf(td * fs);
      ev[9 + i]  = __sinf(ta * fs); ev[13 + i] = __cosf(ta * fs);
      ev[17 + i] = __sinf(te * fs); ev[21 + i] = __cosf(te * fs);
    }
  }
  __syncthreads();

  // ================= KEY branch =================
  buildIN(IN, features, PK, gr0, 0, tid);
  __syncthreads();
  gemm_layer<3, 16, true >(IN, 96,  (const _Float16*)(ws + OFF_WK0F), WBUF, BIASS + 0,    H1, 256, tid);
  __syncthreads();
  gemm_layer<8, 16, true >(H1, 256, (const _Float16*)(ws + OFF_WK1F), WBUF, BIASS + 256,  H2, 256, tid);
  __syncthreads();
  gemm_layer<8, 8,  false>(H2, 256, (const _Float16*)(ws + OFF_WCKF), WBUF, BIASS + 512,  KH, 128, tid);
  __syncthreads();

  // ================= VALUE branch =================
  buildIN(IN, features, PV, gr0, 1, tid);
  __syncthreads();
  gemm_layer<3, 16, true >(IN, 96,  (const _Float16*)(ws + OFF_WV0F), WBUF, BIASS + 640,  H1, 256, tid);
  __syncthreads();
  gemm_layer<8, 16, true >(H1, 256, (const _Float16*)(ws + OFF_WV1F), WBUF, BIASS + 896,  H2, 256, tid);
  __syncthreads();
  gemm_layer<8, 8,  false>(H2, 256, (const _Float16*)(ws + OFF_WCVF), WBUF, BIASS + 1152, VH, 128, tid);
  __syncthreads();

  // ================= attention (8 heads x 32 tokens per ray) =================
  {
    int wid = tid >> 5, lane = tid & 31;
#pragma unroll
    for (int i = 0; i < 4; ++i) {                 // 4 (ray,head) pairs per wave
      int p  = wid * 4 + i;
      int rr = p >> 3, hh = p & 7;
      const _Float16* kr = KH + (size_t)(rr * 32 + lane) * 128 + hh * 16;
      const float*    qp = QSH + rr * 128 + hh * 16;
      float x = 0.f;
#pragma unroll
      for (int d2 = 0; d2 < 16; ++d2) x += qp[d2] * (float)kr[d2];
      float m = x;
#pragma unroll
      for (int off = 16; off >= 1; off >>= 1) m = fmaxf(m, __shfl_xor(m, off, 32));
      float e = __expf(x - m);
      float s = e;
#pragma unroll
      for (int off = 16; off >= 1; off >>= 1) s += __shfl_xor(s, off, 32);
      ATTN[p * 32 + lane] = e / s;
    }
  }
  __syncthreads();

  // attn-weight output (mean over heads)
  if (tid < 128) {
    int rr = tid >> 5, l = tid & 31;
    float s = 0.f;
#pragma unroll
    for (int hh = 0; hh < 8; ++hh) s += ATTN[(rr * 8 + hh) * 32 + l];
    out[OUT_ATTN_OFF + (size_t)(gr0 + rr) * 32 + l] = s * 0.125f;
  }
  // outh = attn @ vh
#pragma unroll
  for (int rep = 0; rep < 2; ++rep) {
    int j = rep * 256 + tid;
    int rr = j >> 7, e = j & 127, hh = e >> 4;
    const float* at = ATTN + (rr * 8 + hh) * 32;
    float o = 0.f;
#pragma unroll
    for (int l = 0; l < 32; ++l) o += at[l] * (float)VH[(size_t)(rr * 32 + l) * 128 + e];
    OUTH[j] = o;
  }
  __syncthreads();
  // out = outh @ (Wo@Wout) + boc
#pragma unroll
  for (int rep = 0; rep < 2; ++rep) {
    int j = rep * 256 + tid;
    int rr = j >> 7, eo = j & 127;
    const float* oh = OUTH + rr * 128;
    float a = boc[eo];
    for (int e = 0; e < 128; ++e) a += oh[e] * woo[e * 128 + eo];
    out[(size_t)(gr0 + rr) * 128 + eo] = a;
  }
}

// ===========================================================================
extern "C" void kernel_launch(void* const* d_in, const int* in_sizes, int n_in,
                              void* d_out, int out_size, void* d_ws, size_t ws_size,
                              hipStream_t stream) {
  (void)in_sizes; (void)n_in; (void)out_size; (void)ws_size;
  const float* dirs     = (const float*)d_in[0];
  const float* feats    = (const float*)d_in[1];
  const float* distance = (const float*)d_in[2];
  /* d_in[3] projected_distance unused (use_projected=False) */
  const float* pitch    = (const float*)d_in[4];
  const float* azimuth  = (const float*)d_in[5];
  const float* txd      = (const float*)d_in[6];
  const float* txa      = (const float*)d_in[7];
  const float* txe      = (const float*)d_in[8];
  const float* Wk0 = (const float*)d_in[9];   const float* bk0 = (const float*)d_in[10];
  const float* Wk1 = (const float*)d_in[11];  const float* bk1 = (const float*)d_in[12];
  const float* Wv0 = (const float*)d_in[13];  const float* bv0 = (const float*)d_in[14];
  const float* Wv1 = (const float*)d_in[15];  const float* bv1 = (const float*)d_in[16];
  const float* Wkey = (const float*)d_in[17]; const float* bkey = (const float*)d_in[18];
  const float* Wval = (const float*)d_in[19]; const float* bval = (const float*)d_in[20];
  const float* Wr0 = (const float*)d_in[21];  const float* br0 = (const float*)d_in[22];
  const float* Wqr = (const float*)d_in[23];  const float* bqr = (const float*)d_in[24];
  const float* Wq  = (const float*)d_in[25];  const float* bq  = (const float*)d_in[26];
  const float* Wka = (const float*)d_in[27];  const float* bka = (const float*)d_in[28];
  const float* Wva = (const float*)d_in[29];  const float* bva = (const float*)d_in[30];
  const float* Wo  = (const float*)d_in[31];  const float* bo  = (const float*)d_in[32];
  const float* Wout = (const float*)d_in[33]; const float* bout = (const float*)d_in[34];
  char*  ws  = (char*)d_ws;
  float* out = (float*)d_out;

  k_combine<<<322, 256, 0, stream>>>(Wkey, bkey, Wka, bka, Wval, bval, Wva, bva,
                                     Wo, bo, Wout, bout, ws);
  k_pack<<<480, 256, 0, stream>>>(Wk0, Wv0, Wk1, Wv1, ws);
  k_query<<<N_RAYS, 128, 0, stream>>>(dirs, Wr0, br0, Wqr, bqr, Wq, bq, ws);
  k_main<<<N_RAYS / 4, 256, SMEM_MAIN, stream>>>(feats, distance, pitch, azimuth,
                                                 txd, txa, txe, bk0, bk1, bv0, bv1,
                                                 ws, out);
}
